// TemporalEmbedNN_16965120819455
// MI455X (gfx1250) — matrix operand order repaired and checked by reading, not compiled, not run on hardware
//
#include <hip/hip_runtime.h>

// ---------------------------------------------------------------------------
// TemporalEmbedNN for MI455X (gfx1250, wave32, WMMA).
// Encoder MLP [160->512->512->32] over 327,680 rows dominates (~236 GFLOP):
// run all GEMMs as v_wmma_f32_16x16x32_bf16 with f32 accumulate; weights are
// converted to bf16 once (1.4 MB, L2-resident); LN / bias / leaky in f32.
// ---------------------------------------------------------------------------

#define BATCH 256
#define CTX   20
#define NN    64
#define NC    128
#define EMB   32
#define OUTL  32
#define HID   512
#define NEG_SLOPE 0.01f
#define LN_EPS    1e-5f

typedef __bf16 bf16;
typedef __attribute__((ext_vector_type(16))) __bf16 bf16x16;
typedef __attribute__((ext_vector_type(8)))  float  f32x8;

struct alignas(16) u32x4 { unsigned int x, y, z, w; };

// Build a 16-element bf16 fragment from two 16-byte chunks (-> b128 loads).
static __device__ __forceinline__ bf16x16 load_frag16(const bf16* p0, const bf16* p1) {
  union { bf16x16 v; u32x4 h[2]; } u;
  u.h[0] = *(const u32x4*)p0;
  u.h[1] = *(const u32x4*)p1;
  return u.v;
}

// Row-wise LayerNorm over `dim` columns of a [64 x stride] bf16 LDS buffer.
// 4 threads per row compute partial sums; reduce; normalize with gamma/beta.
static __device__ __forceinline__
void layer_norm(const bf16* s_in, int stride, int dim,
                const float* __restrict__ gamma, const float* __restrict__ beta,
                bf16* s_out, float* s_part, float* s_stats, int tid)
{
  __syncthreads();                       // input buffer fully written
  const int row = tid >> 2, q = tid & 3, per = dim >> 2;
  const bf16* r = s_in + row * stride + q * per;
  float s = 0.f, ss = 0.f;
  for (int i = 0; i < per; ++i) { float x = (float)r[i]; s += x; ss += x * x; }
  s_part[2 * tid]     = s;
  s_part[2 * tid + 1] = ss;
  __syncthreads();
  if (tid < 64) {
    float sm = 0.f, sq = 0.f;
    for (int k = 0; k < 4; ++k) {
      sm += s_part[2 * (4 * tid + k)];
      sq += s_part[2 * (4 * tid + k) + 1];
    }
    float mean = sm / dim;
    float var  = sq / dim - mean * mean;     // population variance (ddof=0)
    s_stats[2 * tid]     = mean;
    s_stats[2 * tid + 1] = rsqrtf(var + LN_EPS);
  }
  __syncthreads();
  const float mean = s_stats[2 * row], rstd = s_stats[2 * row + 1];
  bf16* o = s_out + row * stride + q * per;
  for (int i = 0; i < per; ++i) {
    float x = (float)r[i];
    o[i] = (bf16)(((x - mean) * rstd) * gamma[q * per + i] + beta[q * per + i]);
  }
  __syncthreads();
}

// One Linear layer: out[64 x NOUT] = in[64 x K] * W^T + bias (+leaky).
// 8 waves; each task = 16x32 output tile (2 WMMA accumulators share A frag).
// A-frag layout (ISA 7.12.2, 16-bit A 16x32): lane m=L&15, hi=L>>4;
//   VGPR0-3 hold K = 8*hi + 0..7, VGPR4-7 hold K = 16 + 8*hi + 0..7.
// B-frag (32x16): lane n=L&15, hi=L>>4; same K layout as A (VGPR0-3: 8*hi+0..7, VGPR4-7: 16+8*hi+0..7).
// W is row-major [NOUT][K] so both halves are contiguous 16B chunks.
static __device__ __forceinline__
void gemm_layer(const bf16* s_in, int in_stride,
                const bf16* __restrict__ wmat, int K, int NOUT,
                const float* __restrict__ bias,
                bf16* out, long out_stride,
                bool leaky, int tid)
{
  const int wave = tid >> 5, lane = tid & 31;
  const int nl = lane & 15, hi = lane >> 4;
  const int tasks = NOUT >> 3;                 // 4 m-tiles * (NOUT/32) pairs
  for (int t = wave; t < tasks; t += 8) {      // wave-uniform: EXEC stays full
    const int mt = t & 3;
    const int n0 = (t >> 2) << 5;
    const int n1 = n0 + 16;
    f32x8 c0 = {0.f,0.f,0.f,0.f,0.f,0.f,0.f,0.f};
    f32x8 c1 = {0.f,0.f,0.f,0.f,0.f,0.f,0.f,0.f};
    const bf16* arow = s_in + (mt * 16 + nl) * in_stride + 8 * hi;
    const bf16* w0   = wmat + (size_t)(n0 + nl) * K + 8 * hi;
    const bf16* w1   = wmat + (size_t)(n1 + nl) * K + 8 * hi;
    for (int kc = 0; kc < K; kc += 32) {
      bf16x16 a  = load_frag16(arow + kc, arow + kc + 16);
      bf16x16 b0 = load_frag16(w0 + kc,   w0 + kc + 16);
      bf16x16 b1 = load_frag16(w1 + kc,   w1 + kc + 16);
      c0 = __builtin_amdgcn_wmma_f32_16x16x32_bf16(false, a, false, b0,
                                                   (short)0, c0, false, false);
      c1 = __builtin_amdgcn_wmma_f32_16x16x32_bf16(false, a, false, b1,
                                                   (short)0, c1, false, false);
    }
    const float bi0 = bias[n0 + nl], bi1 = bias[n1 + nl];
    // C/D layout: lane n = L&15; VGPR j -> row m = 8*hi + j.
#pragma unroll
    for (int j = 0; j < 8; ++j) {
      const int m = mt * 16 + hi * 8 + j;
      float v0 = c0[j] + bi0;
      float v1 = c1[j] + bi1;
      if (leaky) {
        v0 = v0 > 0.f ? v0 : NEG_SLOPE * v0;
        v1 = v1 > 0.f ? v1 : NEG_SLOPE * v1;
      }
      out[(long)m * out_stride + n0 + nl] = (bf16)v0;
      out[(long)m * out_stride + n1 + nl] = (bf16)v1;
    }
  }
}

// ---------------------------------------------------------------------------
// Kernel 1: encoder MLP. One block = one (b,l) pair = 64 node-rows.
// ---------------------------------------------------------------------------
__global__ __launch_bounds__(256)
void encoder_kernel(const float* __restrict__ samples,
                    const float* __restrict__ gmask,
                    const float* __restrict__ emb,
                    const float* g0, const float* be0, const float* b0, const bf16* w0,
                    const float* g1, const float* be1, const float* b1, const bf16* w1,
                    const float* g2, const float* be2, const float* b2, const bf16* w2,
                    bf16* __restrict__ enc)
{
  __shared__ alignas(16) bf16 s_raw[64 * 520];
  __shared__ alignas(16) bf16 s_nrm[64 * 520];
  __shared__ float s_part[512];
  __shared__ float s_stats[128];
  __shared__ float s_smp[NC];

  const int tid = threadIdx.x;
  const int bl  = blockIdx.x;          // b*CTX + l
  const int l   = bl % CTX;

  if (tid < NC) s_smp[tid] = samples[(size_t)bl * NC + tid];
  __syncthreads();

  // Stage 0: raw input row n = [samples * group_mask[n] (128) | emb[l,n] (32)]
  {
    const int n = tid >> 2, q = tid & 3;        // 40 cols per thread
    for (int i = 0; i < 40; ++i) {
      const int c = q * 40 + i;
      float v = (c < NC) ? s_smp[c] * gmask[n * NC + c]
                         : emb[((size_t)l * NN + n) * EMB + (c - NC)];
      s_raw[n * 520 + c] = (bf16)v;
    }
  }

  layer_norm(s_raw, 520, NC + EMB, g0, be0, s_nrm, s_part, s_stats, tid);
  gemm_layer(s_nrm, 520, w0, NC + EMB, HID, b0, s_raw, 520, true, tid);
  layer_norm(s_raw, 520, HID, g1, be1, s_nrm, s_part, s_stats, tid);
  gemm_layer(s_nrm, 520, w1, HID, HID, b1, s_raw, 520, true, tid);
  layer_norm(s_raw, 520, HID, g2, be2, s_nrm, s_part, s_stats, tid);
  // Final layer: write encoded[b,l,n,o] (bf16) straight to global.
  gemm_layer(s_nrm, 520, w2, HID, OUTL, b2,
             enc + (size_t)bl * NN * OUTL, OUTL, false, tid);
}

// ---------------------------------------------------------------------------
// Kernel 2: agg[b,j,o] = sum_{l,i} (w*graphs)[l,i,j] * enc[b,l,i,o]  (f32 VALU;
// only ~0.7 GMAC total). One block per batch b.
// ---------------------------------------------------------------------------
__global__ __launch_bounds__(256)
void aggregate_kernel(const float* __restrict__ wmat,
                      const float* __restrict__ graphs,
                      const bf16* __restrict__ enc,
                      float* __restrict__ agg)
{
  __shared__ float s_wg[NN * NN];    // 16 KB
  __shared__ float s_e[NN * OUTL];   //  8 KB
  const int tid = threadIdx.x;
  const int b = blockIdx.x;
  const int o = tid & 31, jg = tid >> 5;
  float acc[8];
#pragma unroll
  for (int r = 0; r < 8; ++r) acc[r] = 0.f;

  for (int l = 0; l < CTX; ++l) {
    for (int i = 0; i < 16; ++i) {
      const int x = i * 256 + tid;
      s_wg[x] = wmat[(size_t)l * NN * NN + x] * graphs[(size_t)l * NN * NN + x];
    }
    const bf16* e = enc + ((size_t)b * CTX + l) * NN * OUTL;
    for (int i = 0; i < 8; ++i) {
      const int x = i * 256 + tid;
      s_e[x] = (float)e[x];
    }
    __syncthreads();
    for (int i = 0; i < NN; ++i) {
      const float ev = s_e[i * OUTL + o];
#pragma unroll
      for (int r = 0; r < 8; ++r)
        acc[r] += s_wg[i * NN + jg + 8 * r] * ev;   // broadcast read, no conflicts
    }
    __syncthreads();
  }
#pragma unroll
  for (int r = 0; r < 8; ++r)
    agg[((size_t)b * NN + jg + 8 * r) * OUTL + o] = acc[r];
}

// ---------------------------------------------------------------------------
// Kernel 3: decoder MLP + final group-mask contraction. One block per b.
// ---------------------------------------------------------------------------
__global__ __launch_bounds__(256)
void decoder_kernel(const float* __restrict__ agg,
                    const float* __restrict__ emb,
                    const float* __restrict__ gmask,
                    const float* g0, const float* be0, const float* b0, const bf16* w0,
                    const float* g1, const float* be1, const float* b1, const bf16* w1,
                    const float* g2, const float* be2, const float* b2, const bf16* w2,
                    float* __restrict__ outp)
{
  __shared__ alignas(16) bf16 s_raw[64 * 520];
  __shared__ alignas(16) bf16 s_nrm[64 * 520];
  __shared__ float s_part[512];
  __shared__ float s_stats[128];

  const int tid = threadIdx.x;
  const int b = blockIdx.x;

  // Stage 0: row n = [agg[b,n] (32) | emb[last,n] (32)]
  {
    const int n = tid >> 2, q = tid & 3;        // 16 cols per thread
    for (int i = 0; i < 16; ++i) {
      const int c = q * 16 + i;
      float v = (c < OUTL) ? agg[((size_t)b * NN + n) * OUTL + c]
                           : emb[((size_t)(CTX - 1) * NN + n) * EMB + (c - OUTL)];
      s_raw[n * 520 + c] = (bf16)v;
    }
  }

  layer_norm(s_raw, 520, OUTL + EMB, g0, be0, s_nrm, s_part, s_stats, tid);
  gemm_layer(s_nrm, 520, w0, OUTL + EMB, HID, b0, s_raw, 520, true, tid);
  layer_norm(s_raw, 520, HID, g1, be1, s_nrm, s_part, s_stats, tid);
  gemm_layer(s_nrm, 520, w1, HID, HID, b1, s_raw, 520, true, tid);
  layer_norm(s_raw, 520, HID, g2, be2, s_nrm, s_part, s_stats, tid);
  gemm_layer(s_nrm, 520, w2, HID, NC, b2, s_raw, 520, false, tid);
  __syncthreads();

  // out[b,j] = sum_n decoded[n,j] * gmask[n,j]
  if (tid < NC) {
    float acc = 0.f;
    for (int n = 0; n < NN; ++n)
      acc += (float)s_raw[n * 520 + tid] * gmask[n * NC + tid];
    outp[(size_t)b * NC + tid] = acc;
  }
}

// ---------------------------------------------------------------------------
// Kernel 0: fp32 -> bf16 weight conversion into workspace.
// ---------------------------------------------------------------------------
__global__ void cvt_bf16_kernel(const float* __restrict__ s, bf16* __restrict__ d, int n) {
  int i = blockIdx.x * blockDim.x + threadIdx.x;
  if (i < n) d[i] = (bf16)s[i];
}

extern "C" void kernel_launch(void* const* d_in, const int* in_sizes, int n_in,
                              void* d_out, int out_size, void* d_ws, size_t ws_size,
                              hipStream_t stream) {
  // setup_inputs() order with param dicts pytree-flattened (keys sorted:
  // b[0..2], ln_b[0..2], ln_g[0..2], w[0..2]).
  const float* samples = (const float*)d_in[0];
  const float* graphs  = (const float*)d_in[1];
  const float* gmask   = (const float*)d_in[2];
  const float* emb     = (const float*)d_in[3];
  const float* wmat    = (const float*)d_in[4];
  const float* l_b[3]   = {(const float*)d_in[5],  (const float*)d_in[6],  (const float*)d_in[7]};
  const float* l_lnb[3] = {(const float*)d_in[8],  (const float*)d_in[9],  (const float*)d_in[10]};
  const float* l_lng[3] = {(const float*)d_in[11], (const float*)d_in[12], (const float*)d_in[13]};
  const float* l_w[3]   = {(const float*)d_in[14], (const float*)d_in[15], (const float*)d_in[16]};
  const float* z_b[3]   = {(const float*)d_in[17], (const float*)d_in[18], (const float*)d_in[19]};
  const float* z_lnb[3] = {(const float*)d_in[20], (const float*)d_in[21], (const float*)d_in[22]};
  const float* z_lng[3] = {(const float*)d_in[23], (const float*)d_in[24], (const float*)d_in[25]};
  const float* z_w[3]   = {(const float*)d_in[26], (const float*)d_in[27], (const float*)d_in[28]};

  // Workspace layout (bytes), all 16B-aligned:
  char* ws = (char*)d_ws;
  bf16*  lw0 = (bf16*)(ws + 0);          // 512*160*2 = 163840
  bf16*  lw1 = (bf16*)(ws + 163840);     // 512*512*2 = 524288
  bf16*  lw2 = (bf16*)(ws + 688128);     //  32*512*2 =  32768
  bf16*  zw0 = (bf16*)(ws + 720896);     // 512* 64*2 =  65536
  bf16*  zw1 = (bf16*)(ws + 786432);     // 512*512*2 = 524288
  bf16*  zw2 = (bf16*)(ws + 1310720);    // 128*512*2 = 131072
  bf16*  enc = (bf16*)(ws + 1441792);    // 256*20*64*32*2 = 20971520
  float* agg = (float*)(ws + 22413312);  // 256*64*32*4    =  2097152

  auto cvt = [&](const float* s, bf16* d, int n) {
    cvt_bf16_kernel<<<(n + 255) / 256, 256, 0, stream>>>(s, d, n);
  };
  cvt(l_w[0], lw0, 512 * 160);
  cvt(l_w[1], lw1, 512 * 512);
  cvt(l_w[2], lw2, 32 * 512);
  cvt(z_w[0], zw0, 512 * 64);
  cvt(z_w[1], zw1, 512 * 512);
  cvt(z_w[2], zw2, 128 * 512);

  encoder_kernel<<<BATCH * CTX, 256, 0, stream>>>(
      samples, gmask, emb,
      l_lng[0], l_lnb[0], l_b[0], lw0,
      l_lng[1], l_lnb[1], l_b[1], lw1,
      l_lng[2], l_lnb[2], l_b[2], lw2, enc);

  aggregate_kernel<<<BATCH, 256, 0, stream>>>(wmat, graphs, enc, agg);

  decoder_kernel<<<BATCH, 256, 0, stream>>>(
      agg, emb, gmask,
      z_lng[0], z_lnb[0], z_b[0], zw0,
      z_lng[1], z_lnb[1], z_b[1], zw1,
      z_lng[2], z_lnb[2], z_b[2], zw2, (float*)d_out);

  (void)in_sizes; (void)n_in; (void)out_size; (void)ws_size;
}
